// BondRefine_19911468384606
// MI455X (gfx1250) — matrix operand order repaired and checked by reading, not compiled, not run on hardware
//
#include <hip/hip_runtime.h>
#include <hip/hip_fp16.h>

typedef __attribute__((ext_vector_type(16))) _Float16 v16h;
typedef __attribute__((ext_vector_type(8)))  float    v8f;

#define G_   256
#define DN   64
#define DE   32
#define KPAD 192
#define EPS_ 1e-5f

// ---- workspace layout (float offsets) ----
// stats region (zeroed every launch):
#define S_X   0          // G*4 : sumX, sumY, sumZ, cntNodes
#define S_H   (G_*4)     // G*2 : sumH, sumHH
#define S_E   (G_*6)     // G*3 : sumE, sumEE, cntE
#define S_B   (G_*9)     // G*2 : sumB, sumBB
#define S_XM  (G_*11)    // G*4 : xmean(3) + pad
#define S_HS  (G_*15)    // G*2 : hmean, hinv
#define S_ES  (G_*17)    // G*2 : emean, einv
#define S_BS  (G_*19)    // G*2 : bmean, binv
#define STATS_F 8192

union F16x16 { uint4 u[2]; v16h h; };
union F16x4  { _Float16 h[4]; uint2 u; };

// ---------------- zero stats ----------------
__global__ __launch_bounds__(256) void k_zero(float* stats) {
  for (int i = blockIdx.x * blockDim.x + threadIdx.x; i < STATS_F; i += gridDim.x * blockDim.x)
    stats[i] = 0.f;
}

// ---------------- node stats: X sums + H scalar sums ----------------
__global__ __launch_bounds__(256) void k_node_stats(const int* __restrict__ batch,
                                                    const float* __restrict__ X,
                                                    const float* __restrict__ H,
                                                    float* stats, int N) {
  __shared__ float bins[G_ * 6];
  for (int i = threadIdx.x; i < G_ * 6; i += blockDim.x) bins[i] = 0.f;
  __syncthreads();
  int n = blockIdx.x * blockDim.x + threadIdx.x;
  if (n < N) {
    int g = batch[n];
    float sh = 0.f, shh = 0.f;
    #pragma unroll
    for (int j = 0; j < DN; j += 4) {
      float4 v = *(const float4*)(H + (size_t)n * DN + j);
      sh  += v.x + v.y + v.z + v.w;
      shh += v.x * v.x + v.y * v.y + v.z * v.z + v.w * v.w;
    }
    atomicAdd(&bins[g * 6 + 0], X[n * 3 + 0]);
    atomicAdd(&bins[g * 6 + 1], X[n * 3 + 1]);
    atomicAdd(&bins[g * 6 + 2], X[n * 3 + 2]);
    atomicAdd(&bins[g * 6 + 3], 1.f);
    atomicAdd(&bins[g * 6 + 4], sh);
    atomicAdd(&bins[g * 6 + 5], shh);
  }
  __syncthreads();
  for (int i = threadIdx.x; i < G_ * 6; i += blockDim.x) {
    float v = bins[i];
    if (v != 0.f) {
      int g = i / 6, c = i % 6;
      float* dst = (c < 4) ? &stats[S_X + g * 4 + c] : &stats[S_H + g * 2 + (c - 4)];
      atomicAdd(dst, v);
    }
  }
}

// ---------------- edge stats: edge_attr scalar sums + counts + edge_batch ----------------
__global__ __launch_bounds__(256) void k_edge_stats(const int* __restrict__ ei,
                                                    const int* __restrict__ batch,
                                                    const float* __restrict__ EA,
                                                    float* stats, int* __restrict__ eb, int E) {
  __shared__ float bins[G_ * 3];
  for (int i = threadIdx.x; i < G_ * 3; i += blockDim.x) bins[i] = 0.f;
  __syncthreads();
  int e = blockIdx.x * blockDim.x + threadIdx.x;
  if (e < E) {
    int s = ei[e];              // edge_index[0][e]
    int g = batch[s];
    eb[e] = g;
    float se = 0.f, see = 0.f;
    #pragma unroll
    for (int j = 0; j < DE; j += 4) {
      float4 v = *(const float4*)(EA + (size_t)e * DE + j);
      se  += v.x + v.y + v.z + v.w;
      see += v.x * v.x + v.y * v.y + v.z * v.z + v.w * v.w;
    }
    atomicAdd(&bins[g * 3 + 0], se);
    atomicAdd(&bins[g * 3 + 1], see);
    atomicAdd(&bins[g * 3 + 2], 1.f);
  }
  __syncthreads();
  for (int i = threadIdx.x; i < G_ * 3; i += blockDim.x) {
    float v = bins[i];
    if (v != 0.f) atomicAdd(&stats[S_E + i], v);
  }
}

// ---------------- per-graph finalize: means + inv std ----------------
__global__ __launch_bounds__(256) void k_graph_stats(float* stats) {
  int g = threadIdx.x;
  float cn = stats[S_X + g * 4 + 3];
  float cd = fmaxf(cn, 1.f);
  stats[S_XM + g * 4 + 0] = stats[S_X + g * 4 + 0] / cd;
  stats[S_XM + g * 4 + 1] = stats[S_X + g * 4 + 1] / cd;
  stats[S_XM + g * 4 + 2] = stats[S_X + g * 4 + 2] / cd;

  float dh = fmaxf(cn * (float)DN, 1.f);
  float hm = stats[S_H + g * 2 + 0] / dh;
  float hv = stats[S_H + g * 2 + 1] / dh - hm * hm;
  stats[S_HS + g * 2 + 0] = hm;
  stats[S_HS + g * 2 + 1] = rsqrtf(fmaxf(hv, 0.f) + EPS_);

  float ce = stats[S_E + g * 3 + 2];
  float de = fmaxf(ce * (float)DE, 1.f);
  float em = stats[S_E + g * 3 + 0] / de;
  float ev = stats[S_E + g * 3 + 1] / de - em * em;
  stats[S_ES + g * 2 + 0] = em;
  stats[S_ES + g * 2 + 1] = rsqrtf(fmaxf(ev, 0.f) + EPS_);
}

// ---------------- node prep: Xc (f32) + normalized Hn (f16) ----------------
__global__ __launch_bounds__(256) void k_node_prep(const int* __restrict__ batch,
                                                   const float* __restrict__ X,
                                                   const float* __restrict__ H,
                                                   const float* __restrict__ h_gamma,
                                                   const float* __restrict__ h_beta,
                                                   const float* __restrict__ stats,
                                                   float* __restrict__ Xc,
                                                   _Float16* __restrict__ Hn, int N) {
  int idx = blockIdx.x * blockDim.x + threadIdx.x;
  int n = idx >> 4;
  if (n >= N) return;
  int j4 = (idx & 15) * 4;
  int g = batch[n];
  float hm = stats[S_HS + g * 2 + 0];
  float hi = stats[S_HS + g * 2 + 1];
  float4 h = *(const float4*)(H + (size_t)n * DN + j4);
  F16x4 o;
  o.h[0] = (_Float16)((h.x - hm) * hi * h_gamma[j4 + 0] + h_beta[j4 + 0]);
  o.h[1] = (_Float16)((h.y - hm) * hi * h_gamma[j4 + 1] + h_beta[j4 + 1]);
  o.h[2] = (_Float16)((h.z - hm) * hi * h_gamma[j4 + 2] + h_beta[j4 + 2]);
  o.h[3] = (_Float16)((h.w - hm) * hi * h_gamma[j4 + 3] + h_beta[j4 + 3]);
  *(uint2*)(Hn + (size_t)n * DN + j4) = o.u;
  int c = idx & 15;
  if (c < 3) Xc[n * 4 + c] = X[n * 3 + c] - stats[S_XM + g * 4 + c];
}

// ---------------- main WMMA edge MLP ----------------
// infeats column remap (same permutation applied to W1 rows, dot product invariant):
//   k 0..63   = Hn[tgt]            (W1 rows 0..63)
//   k 64..127 = Hn[src]            (W1 rows 64..127)
//   k 128..159= normalized ea      (W1 rows 129..160)  [16B aligned]
//   k 160     = rel_dist           (W1 row 128)
//   k 161..191= zero padding       (W1 zero)
//
// Hn rows are gathered with GLOBAL_LOAD_ASYNC_TO_LDS_B128: the ISA adds the
// same INST_OFFSET to both the LDS dest VGPR and the global addr VGPR, and our
// chunk stride is +16B in both spaces, so 8 instructions with offset:0..112
// move one 128B row without any VGPR staging. Tracked by ASYNCcnt.
#define ASYNC16(OFF) \
  asm volatile("global_load_async_to_lds_b128 %0, %1, off offset:" #OFF \
               :: "v"(lds0), "v"(ga) : "memory")

__global__ __launch_bounds__(256) void k_edge_mlp(
    const int* __restrict__ ei, const int* __restrict__ batch,
    const float* __restrict__ EA,
    const float* __restrict__ e_gamma, const float* __restrict__ e_beta,
    const float* __restrict__ W1, const float* __restrict__ b1,
    const float* __restrict__ W2, const float* __restrict__ b2,
    const float* __restrict__ stats, const float* __restrict__ Xc,
    const _Float16* __restrict__ Hn,
    float* __restrict__ out, int E, int Ntiles) {
  __shared__ _Float16 sW1[DE * KPAD];      // W1^T [n][k], f16, 12 KB
  __shared__ _Float16 sW2[DE * DE];        // W2^T [n][k], f16, 2 KB
  __shared__ float sb1[DE], sb2[DE], sEG[DE], sEB[DE];
  __shared__ _Float16 sIF[8 * 16 * KPAD];  // per-wave 16x192 infeats, 48 KB

  int tid = threadIdx.x;
  for (int i = tid; i < DE * KPAD; i += 256) {
    int n = i / KPAD, k = i % KPAD;
    float v = 0.f;
    if (k < 128)      v = W1[(size_t)k * DE + n];
    else if (k < 160) v = W1[(size_t)(129 + (k - 128)) * DE + n];
    else if (k == 160) v = W1[(size_t)128 * DE + n];
    sW1[i] = (_Float16)v;
  }
  for (int i = tid; i < DE * DE; i += 256) {
    int n = i / DE, k = i % DE;
    sW2[i] = (_Float16)W2[(size_t)k * DE + n];
  }
  if (tid < DE) { sb1[tid] = b1[tid]; sb2[tid] = b2[tid]; sEG[tid] = e_gamma[tid]; sEB[tid] = e_beta[tid]; }
  __syncthreads();

  int wave = tid >> 5, lane = tid & 31;
  _Float16* IF = sIF + wave * 16 * KPAD;

  // zero the pad columns once (never overwritten inside the tile loop)
  if (lane < 16)
    for (int k = 161; k < KPAD; ++k) IF[lane * KPAD + k] = (_Float16)0.f;

  int nn = lane & 15;
  // preload W2 B-fragments (N tiles 0..15 and 16..31), kept across tiles
  F16x16 w2f0, w2f1;
  {
    int kk = (lane >> 4) << 4;  // lanes 0-15: K 0..15, lanes 16-31: K 16..31
    w2f0.u[0] = *(const uint4*)(sW2 + nn * DE + kk);
    w2f0.u[1] = *(const uint4*)(sW2 + nn * DE + kk + 8);
    w2f1.u[0] = *(const uint4*)(sW2 + (nn + 16) * DE + kk);
    w2f1.u[1] = *(const uint4*)(sW2 + (nn + 16) * DE + kk + 8);
  }

  int r  = lane & 15;
  int ao = (lane >> 4) << 4;   // A frag: lanes 0-15 -> K 0..15, lanes 16-31 -> K 16..31
  int bo = (lane >> 4) << 4;   // B frag: lanes 0-15 -> K 0..15, lanes 16-31 -> K 16..31

  // persistent per-wave tile loop
  for (int tile = blockIdx.x * 8 + wave; tile < Ntiles; tile += gridDim.x * 8) {
    // ---- gather: 2 lanes per edge ----
    {
      int el = lane >> 1, hf = lane & 1;
      int e = tile * 16 + el;
      if (e < E) {
        int src = ei[e], tgt = ei[E + e];
        int node = hf ? src : tgt;
        // async-copy one 128B Hn row straight into LDS (no VGPR staging)
        unsigned lds0 = (unsigned)(size_t)(IF + el * KPAD + hf * 64);
        unsigned long long ga = (unsigned long long)(const void*)(Hn + (size_t)node * DN);
        ASYNC16(0);  ASYNC16(16); ASYNC16(32); ASYNC16(48);
        ASYNC16(64); ASYNC16(80); ASYNC16(96); ASYNC16(112);
        if (hf == 0) {
          float dx = Xc[src * 4 + 0] - Xc[tgt * 4 + 0];
          float dy = Xc[src * 4 + 1] - Xc[tgt * 4 + 1];
          float dz = Xc[src * 4 + 2] - Xc[tgt * 4 + 2];
          IF[el * KPAD + 160] = (_Float16)(dx * dx + dy * dy + dz * dz);
        } else {
          int g = batch[src];
          float em = stats[S_ES + g * 2 + 0], ev = stats[S_ES + g * 2 + 1];
          #pragma unroll
          for (int j = 0; j < DE; j += 4) {
            float4 v = *(const float4*)(EA + (size_t)e * DE + j);
            F16x4 t;
            t.h[0] = (_Float16)((v.x - em) * ev * sEG[j + 0] + sEB[j + 0]);
            t.h[1] = (_Float16)((v.y - em) * ev * sEG[j + 1] + sEB[j + 1]);
            t.h[2] = (_Float16)((v.z - em) * ev * sEG[j + 2] + sEB[j + 2]);
            t.h[3] = (_Float16)((v.w - em) * ev * sEG[j + 3] + sEB[j + 3]);
            *(uint2*)(IF + el * KPAD + 128 + j) = t.u;
          }
        }
      }
    }
    // async Hn rows must have landed in LDS before fragment reads
    asm volatile("s_wait_asynccnt 0" ::: "memory");
    // (ea / rel_dist DS stores are wave-in-order with the DS loads below)

    // ---- GEMM1: [16 x 192] x [192 x 32] ----
    v8f acc0 = {0.f, 0.f, 0.f, 0.f, 0.f, 0.f, 0.f, 0.f};
    v8f acc1 = {0.f, 0.f, 0.f, 0.f, 0.f, 0.f, 0.f, 0.f};
    #pragma unroll
    for (int ks = 0; ks < 6; ++ks) {
      int k0 = ks * 32;
      F16x16 a, bA, bB;
      a.u[0]  = *(const uint4*)(IF + r * KPAD + k0 + ao);
      a.u[1]  = *(const uint4*)(IF + r * KPAD + k0 + ao + 8);
      bA.u[0] = *(const uint4*)(sW1 + nn * KPAD + k0 + bo);
      bA.u[1] = *(const uint4*)(sW1 + nn * KPAD + k0 + bo + 8);
      bB.u[0] = *(const uint4*)(sW1 + (nn + 16) * KPAD + k0 + bo);
      bB.u[1] = *(const uint4*)(sW1 + (nn + 16) * KPAD + k0 + bo + 8);
      acc0 = __builtin_amdgcn_wmma_f32_16x16x32_f16(false, a.h, false, bA.h, (short)0, acc0, false, false);
      acc1 = __builtin_amdgcn_wmma_f32_16x16x32_f16(false, a.h, false, bB.h, (short)0, acc1, false, false);
    }

    // ---- bias + silu (C layout: lane owns column N, 8 rows) ----
    float bb0 = sb1[nn], bb1 = sb1[nn + 16];
    #pragma unroll
    for (int j = 0; j < 8; ++j) {
      float x0 = acc0[j] + bb0;
      float x1 = acc1[j] + bb1;
      acc0[j] = x0 * __builtin_amdgcn_rcpf(1.f + __expf(-x0));
      acc1[j] = x1 * __builtin_amdgcn_rcpf(1.f + __expf(-x1));
    }

    // ---- re-fragment through LDS (cols 0..31 of IF, pad cols stay zero) ----
    #pragma unroll
    for (int j = 0; j < 8; ++j) {
      int M = j + ((lane >> 4) << 3);
      IF[M * KPAD + nn]      = (_Float16)acc0[j];
      IF[M * KPAD + nn + 16] = (_Float16)acc1[j];
    }

    // ---- GEMM2: [16 x 32] x [32 x 32] ----
    F16x16 a2;
    a2.u[0] = *(const uint4*)(IF + r * KPAD + ao);
    a2.u[1] = *(const uint4*)(IF + r * KPAD + ao + 8);
    v8f h0 = {0.f, 0.f, 0.f, 0.f, 0.f, 0.f, 0.f, 0.f};
    v8f h1 = {0.f, 0.f, 0.f, 0.f, 0.f, 0.f, 0.f, 0.f};
    h0 = __builtin_amdgcn_wmma_f32_16x16x32_f16(false, a2.h, false, w2f0.h, (short)0, h0, false, false);
    h1 = __builtin_amdgcn_wmma_f32_16x16x32_f16(false, a2.h, false, w2f1.h, (short)0, h1, false, false);

    float cb0 = sb2[nn], cb1 = sb2[nn + 16];
    #pragma unroll
    for (int j = 0; j < 8; ++j) {
      int M = j + ((lane >> 4) << 3);
      int e = tile * 16 + M;
      if (e < E) {
        out[(size_t)e * DE + nn]      = h0[j] + cb0;
        out[(size_t)e * DE + nn + 16] = h1[j] + cb1;
      }
    }
  }
}

// ---------------- bond stats over h ----------------
__global__ __launch_bounds__(256) void k_bond_stats(const float* __restrict__ h,
                                                    const int* __restrict__ eb,
                                                    float* stats, int E) {
  __shared__ float bins[G_ * 2];
  for (int i = threadIdx.x; i < G_ * 2; i += blockDim.x) bins[i] = 0.f;
  __syncthreads();
  int e = blockIdx.x * blockDim.x + threadIdx.x;
  if (e < E) {
    int g = eb[e];
    float s = 0.f, ss = 0.f;
    #pragma unroll
    for (int j = 0; j < DE; j += 4) {
      float4 v = *(const float4*)(h + (size_t)e * DE + j);
      s  += v.x + v.y + v.z + v.w;
      ss += v.x * v.x + v.y * v.y + v.z * v.z + v.w * v.w;
    }
    atomicAdd(&bins[g * 2 + 0], s);
    atomicAdd(&bins[g * 2 + 1], ss);
  }
  __syncthreads();
  for (int i = threadIdx.x; i < G_ * 2; i += blockDim.x) {
    float v = bins[i];
    if (v != 0.f) atomicAdd(&stats[S_B + i], v);
  }
}

__global__ __launch_bounds__(256) void k_bond_final(float* stats) {
  int g = threadIdx.x;
  float ce = stats[S_E + g * 3 + 2];
  float d  = fmaxf(ce * (float)DE, 1.f);
  float m  = stats[S_B + g * 2 + 0] / d;
  float v  = stats[S_B + g * 2 + 1] / d - m * m;
  stats[S_BS + g * 2 + 0] = m;
  stats[S_BS + g * 2 + 1] = rsqrtf(fmaxf(v, 0.f) + EPS_);
}

__global__ __launch_bounds__(256) void k_apply(float* __restrict__ out,
                                               const int* __restrict__ eb,
                                               const float* __restrict__ stats,
                                               const float* __restrict__ b_gamma,
                                               const float* __restrict__ b_beta, int E) {
  int idx = blockIdx.x * blockDim.x + threadIdx.x;
  int e = idx >> 3;
  if (e >= E) return;
  int j = (idx & 7) * 4;
  int g = eb[e];
  float m = stats[S_BS + g * 2 + 0], inv = stats[S_BS + g * 2 + 1];
  float4 v = *(float4*)(out + (size_t)e * DE + j);
  v.x = (v.x - m) * inv * b_gamma[j + 0] + b_beta[j + 0];
  v.y = (v.y - m) * inv * b_gamma[j + 1] + b_beta[j + 1];
  v.z = (v.z - m) * inv * b_gamma[j + 2] + b_beta[j + 2];
  v.w = (v.w - m) * inv * b_gamma[j + 3] + b_beta[j + 3];
  *(float4*)(out + (size_t)e * DE + j) = v;
}

extern "C" void kernel_launch(void* const* d_in, const int* in_sizes, int n_in,
                              void* d_out, int out_size, void* d_ws, size_t ws_size,
                              hipStream_t stream) {
  const int*   batch   = (const int*)d_in[0];
  const float* X       = (const float*)d_in[1];
  const float* H       = (const float*)d_in[2];
  const int*   ei      = (const int*)d_in[3];
  const float* EA      = (const float*)d_in[4];
  const float* h_gamma = (const float*)d_in[5];
  const float* h_beta  = (const float*)d_in[6];
  const float* e_gamma = (const float*)d_in[7];
  const float* e_beta  = (const float*)d_in[8];
  const float* b_gamma = (const float*)d_in[9];
  const float* b_beta  = (const float*)d_in[10];
  const float* W1      = (const float*)d_in[11];
  const float* b1      = (const float*)d_in[12];
  const float* W2      = (const float*)d_in[13];
  const float* b2      = (const float*)d_in[14];

  int N = in_sizes[1] / 3;
  int E = in_sizes[4] / DE;

  float*    ws    = (float*)d_ws;
  float*    stats = ws;
  float*    Xc    = ws + STATS_F;                           // N*4 floats
  _Float16* Hn    = (_Float16*)(Xc + (size_t)N * 4);        // N*64 halfs
  int*      eb    = (int*)((char*)Hn + (size_t)N * DN * sizeof(_Float16)); // E ints
  float*    out   = (float*)d_out;

  k_zero<<<32, 256, 0, stream>>>(stats);
  k_node_stats<<<(N + 255) / 256, 256, 0, stream>>>(batch, X, H, stats, N);
  k_edge_stats<<<(E + 255) / 256, 256, 0, stream>>>(ei, batch, EA, stats, eb, E);
  k_graph_stats<<<1, 256, 0, stream>>>(stats);
  k_node_prep<<<(N * 16 + 255) / 256, 256, 0, stream>>>(batch, X, H, h_gamma, h_beta, stats, Xc, Hn, N);

  int Ntiles = (E + 15) / 16;
  int blocks = (Ntiles + 7) / 8;
  if (blocks > 1280) blocks = 1280;      // persistent waves, ~5 tiles each
  k_edge_mlp<<<blocks, 256, 0, stream>>>(ei, batch, EA, e_gamma, e_beta,
                                         W1, b1, W2, b2, stats, Xc, Hn, out, E, Ntiles);

  k_bond_stats<<<(E + 255) / 256, 256, 0, stream>>>(out, eb, stats, E);
  k_bond_final<<<1, 256, 0, stream>>>(stats);
  k_apply<<<(E * 8 + 255) / 256, 256, 0, stream>>>(out, eb, stats, b_gamma, b_beta, E);
}
